// EncoderBlock_24618752540928
// MI455X (gfx1250) — matrix operand order repaired and checked by reading, not compiled, not run on hardware
//
#include <hip/hip_runtime.h>
#include <math.h>

// ---------------------------------------------------------------------------
// CDNA5 (gfx1250) wave32 WMMA encoder block.
// GEMMs run through v_wmma_f32_16x16x32_bf16; tile staging uses the gfx1250
// async global->LDS pipe (global_load_async_to_lds_b128 + s_wait_asynccnt).
// ---------------------------------------------------------------------------

#ifndef USE_ASYNC_LDS
#define USE_ASYNC_LDS 1
#endif

typedef __attribute__((ext_vector_type(16))) __bf16 v16bf;
typedef __attribute__((ext_vector_type(8)))  float  v8f;

#define HEADS 16
#define DKH   64   // head dim

// ---- bf16 <-> f32 helpers (bit-exact RNE, no reliance on native __bf16 ops)
__device__ __forceinline__ __bf16 f2bf(float x) {
  unsigned u = __builtin_bit_cast(unsigned, x);
  unsigned r = u + 0x7FFFu + ((u >> 16) & 1u);
  unsigned short h = (unsigned short)(r >> 16);
  return __builtin_bit_cast(__bf16, h);
}
__device__ __forceinline__ float bf2f(__bf16 b) {
  unsigned short h = __builtin_bit_cast(unsigned short, b);
  return __builtin_bit_cast(float, (unsigned)h << 16);
}

// ---- async global->LDS 16B copy (CDNA5 ASYNCcnt path) ----------------------
__device__ __forceinline__ void stage16(void* lds_dst, const void* gsrc) {
#if USE_ASYNC_LDS
  unsigned loff = (unsigned)(size_t)(__attribute__((address_space(3))) void*)lds_dst;
  unsigned long long g = (unsigned long long)gsrc;
  asm volatile("global_load_async_to_lds_b128 %0, %1, off"
               :: "v"(loff), "v"(g) : "memory");
#else
  *(uint4*)lds_dst = *(const uint4*)gsrc;
#endif
}
__device__ __forceinline__ void stage_fence() {
#if USE_ASYNC_LDS
  asm volatile("s_wait_asynccnt 0x0" ::: "memory");
#endif
}

// ---- WMMA fragment loaders -------------------------------------------------
// A-matrix 16x32 bf16 layout (ISA 7.12.2): lane = M row, half = lane>>4 picks
// K groups {8*half..+7, 16+8*half..+7}; contiguous 16B runs in [M][K] LDS.
__device__ __forceinline__ v16bf load_a_frag(const __bf16* base, int stride,
                                             int m_base, int kk, int lane) {
  const int half = (lane >> 4) & 1;
  const __bf16* p = base + (size_t)(m_base + (lane & 15)) * stride + kk + 8 * half;
  union { v16bf v; uint4 q[2]; } f;
  f.q[0] = *(const uint4*)p;
  f.q[1] = *(const uint4*)(p + 16);
  return f.v;
}
// B-matrix 32x16 bf16: lane = N col, half selects K groups {8*half..+7, 16+8*half..+7},
// contiguous in a K-transposed LDS tile [N][K].
__device__ __forceinline__ v16bf load_b_frag(const __bf16* baseT, int stride,
                                             int n_base, int kk, int lane) {
  const int half = (lane >> 4) & 1;
  const __bf16* p = baseT + (size_t)(n_base + (lane & 15)) * stride + kk + 8 * half;
  union { v16bf v; uint4 q[2]; } f;
  f.q[0] = *(const uint4*)p;
  f.q[1] = *(const uint4*)(p + 16);
  return f.v;
}
__device__ __forceinline__ v8f wmma_bf16(v16bf a, v16bf b, v8f c) {
  return __builtin_amdgcn_wmma_f32_16x16x32_bf16(false, a, false, b,
                                                 (short)0, c, false, false);
}

// ---------------------------------------------------------------------------
// Weight prep: W[K,N] f32 -> WT[N,K] bf16 (tiled transpose through LDS).
// Pre-transposing makes GEMM B-staging a pure 16B-chunk copy (async-friendly).
// ---------------------------------------------------------------------------
__global__ __launch_bounds__(256) void transpose_f32_to_bf16_kernel(
    const float* __restrict__ W, __bf16* __restrict__ WT, int K, int N) {
  __shared__ float t[32][33];
  const int kb = blockIdx.y * 32, nb = blockIdx.x * 32;
  const int tx = threadIdx.x & 31, ty = threadIdx.x >> 5;  // 32x8
#pragma unroll
  for (int i = 0; i < 32; i += 8)
    t[ty + i][tx] = W[(size_t)(kb + ty + i) * N + nb + tx];
  __syncthreads();
#pragma unroll
  for (int i = 0; i < 32; i += 8)
    WT[(size_t)(nb + ty + i) * K + kb + tx] = f2bf(t[tx][ty + i]);
}

// ---------------------------------------------------------------------------
// LayerNorm (torch semantics: unbiased std, alpha*(x-mean)/(std+eps)+beta),
// emits bf16 for downstream WMMA GEMMs. One block per row of D.
// ---------------------------------------------------------------------------
__global__ __launch_bounds__(256) void ln_bf16_kernel(
    const float* __restrict__ x, const float* __restrict__ alpha,
    const float* __restrict__ beta, __bf16* __restrict__ out, int Dd) {
  __shared__ float sh[512];
  const int tid = threadIdx.x;
  const size_t row = blockIdx.x;
  const float* xr = x + row * Dd;
  float s = 0.f, s2 = 0.f;
  for (int i = tid; i < Dd; i += 256) { float v = xr[i]; s += v; s2 += v * v; }
  sh[tid] = s; sh[256 + tid] = s2;
  __syncthreads();
  for (int o = 128; o > 0; o >>= 1) {
    if (tid < o) { sh[tid] += sh[tid + o]; sh[256 + tid] += sh[256 + tid + o]; }
    __syncthreads();
  }
  const float mean = sh[0] / Dd;
  const float var  = (sh[256] - sh[0] * sh[0] / Dd) / (Dd - 1);
  const float inv  = 1.0f / (sqrtf(fmaxf(var, 0.0f)) + 1e-6f);
  for (int i = tid; i < Dd; i += 256)
    out[row * Dd + i] = f2bf(alpha[i] * (xr[i] - mean) * inv + beta[i]);
}

// ---------------------------------------------------------------------------
// Tiled bf16 WMMA GEMM: C[M,N] = epilogue(A[M,K] @ W[K,N]), W given as
// WT[N,K]. Block tile 64x128, BK=32 (one WMMA K-step). 8 waves (2x4), each
// owns 32x32 (2x2 accumulators -> 4 v_wmma per K-step). Both LDS tiles are
// staged with async global->LDS 16B copies (no VGPR round-trip).
// mode: 0 none | 1 relu+1e-6 | 2 bias+relu | 3 +resid | 4 bias+resid
// ---------------------------------------------------------------------------
__global__ __launch_bounds__(256) void gemm_bf16_kernel(
    const __bf16* __restrict__ A, const __bf16* __restrict__ WT,
    const float* __restrict__ bias, const float* __restrict__ resid,
    float* __restrict__ outF, __bf16* __restrict__ outB,
    int M, int N, int K, int mode) {
  __shared__ __align__(16) __bf16 As[64 * 32];    // [m][k]
  __shared__ __align__(16) __bf16 BsT[128 * 32];  // [n][k]
  const int tid  = threadIdx.x;
  const int lane = tid & 31;
  const int wave = tid >> 5;
  const int wm = wave >> 2;      // 0..1
  const int wn = wave & 3;       // 0..3
  const int m0 = blockIdx.y * 64;
  const int n0 = blockIdx.x * 128;

  v8f acc[2][2];
#pragma unroll
  for (int i = 0; i < 2; ++i)
#pragma unroll
    for (int j = 0; j < 2; ++j)
#pragma unroll
      for (int e = 0; e < 8; ++e) acc[i][j][e] = 0.0f;

  const int ar = tid >> 2;        // 0..63
  const int ac = (tid & 3) * 8;   // 0,8,16,24
  const int br = tid >> 1;        // 0..127 (B rows, first half)
  const int bc = (tid & 1) * 8;   // 0,8  (plus +16 via second chunk)
  for (int kk = 0; kk < K; kk += 32) {
    // stage A 64x32: one 16B async chunk per thread
    stage16(As + ar * 32 + ac, A + (size_t)(m0 + ar) * K + kk + ac);
    if (kk + 32 < K)  // gfx1250 global_prefetch_b8 on the next A tile
      __builtin_prefetch(A + (size_t)(m0 + ar) * K + kk + 32 + ac, 0, 1);
    // stage WT 128x32: two 16B async chunks per thread (pure copy, no scatter)
    stage16(BsT + br * 32 + bc,      WT + (size_t)(n0 + br) * K + kk + bc);
    stage16(BsT + br * 32 + bc + 16, WT + (size_t)(n0 + br) * K + kk + bc + 16);
    stage_fence();
    __syncthreads();
    v16bf afr[2], bfr[2];
    afr[0] = load_a_frag(As, 32, wm * 32, 0, lane);
    afr[1] = load_a_frag(As, 32, wm * 32 + 16, 0, lane);
    bfr[0] = load_b_frag(BsT, 32, wn * 32, 0, lane);
    bfr[1] = load_b_frag(BsT, 32, wn * 32 + 16, 0, lane);
#pragma unroll
    for (int i = 0; i < 2; ++i)
#pragma unroll
      for (int j = 0; j < 2; ++j)
        acc[i][j] = wmma_bf16(afr[i], bfr[j], acc[i][j]);
    __syncthreads();
  }

  // epilogue (C layout: VGPR e -> M = e + 8*half, N = lane&15)
#pragma unroll
  for (int i = 0; i < 2; ++i)
#pragma unroll
    for (int j = 0; j < 2; ++j)
#pragma unroll
      for (int e = 0; e < 8; ++e) {
        const int gm = m0 + wm * 32 + i * 16 + e + 8 * (lane >> 4);
        const int gn = n0 + wn * 32 + j * 16 + (lane & 15);
        float v = acc[i][j][e];
        if (mode == 1)      v = fmaxf(v, 0.0f) + 1e-6f;
        else if (mode == 2) v = fmaxf(v + bias[gn], 0.0f);
        else if (mode == 3) v += resid[(size_t)gm * N + gn];
        else if (mode == 4) v += bias[gn] + resid[(size_t)gm * N + gn];
        const size_t o = (size_t)gm * N + gn;
        if (outF) outF[o] = v;
        if (outB) outB[o] = f2bf(v);
      }
}

// ---------------------------------------------------------------------------
// kv[b,h] = sum_n K[b,n,h,:]^T (x) V[b,n,h,:]   (64x64 f32 per head)
// + ksum[b,h] = sum over n,d of K. One block per (b,h); K-loop over tokens.
// Activation transpose -> keep direct LDS staging (scatter) here.
// ---------------------------------------------------------------------------
__global__ __launch_bounds__(256) void linattn_kv_kernel(
    const __bf16* __restrict__ Km, const __bf16* __restrict__ Vm,
    float* __restrict__ kvout, float* __restrict__ ksum, int Ntok, int Dd) {
  __shared__ __align__(16) __bf16 KsT[64 * 32];  // [d][n]
  __shared__ __align__(16) __bf16 VsT[64 * 32];  // [e][n]
  __shared__ float sred[256];
  const int tid = threadIdx.x, lane = tid & 31, wave = tid >> 5;
  const int bh = blockIdx.x, b = bh >> 4, h = bh & 15;
  const __bf16* Kb = Km + (size_t)b * Ntok * Dd + h * DKH;
  const __bf16* Vb = Vm + (size_t)b * Ntok * Dd + h * DKH;
  const int nl = tid & 31;          // token within tile
  const int dg = (tid >> 5) * 8;    // d group
  float ksl = 0.0f;
  v8f acc[2];
#pragma unroll
  for (int u = 0; u < 2; ++u)
#pragma unroll
    for (int e = 0; e < 8; ++e) acc[u][e] = 0.0f;
  const int t0 = wave * 2;          // 16 tiles (4x4), 2 per wave
  for (int n0 = 0; n0 < Ntok; n0 += 32) {
    uint4 kq = *(const uint4*)(Kb + (size_t)(n0 + nl) * Dd + dg);
    uint4 vq = *(const uint4*)(Vb + (size_t)(n0 + nl) * Dd + dg);
    const __bf16* kp = (const __bf16*)&kq;
    const __bf16* vp = (const __bf16*)&vq;
#pragma unroll
    for (int e = 0; e < 8; ++e) {
      KsT[(dg + e) * 32 + nl] = kp[e];
      VsT[(dg + e) * 32 + nl] = vp[e];
      ksl += bf2f(kp[e]);
    }
    __syncthreads();
#pragma unroll
    for (int u = 0; u < 2; ++u) {
      const int t = t0 + u, mi = t >> 2, ni = t & 3;
      v16bf a  = load_a_frag(KsT, 32, mi * 16, 0, lane);
      v16bf bb = load_b_frag(VsT, 32, ni * 16, 0, lane);
      acc[u] = wmma_bf16(a, bb, acc[u]);
    }
    __syncthreads();
  }
#pragma unroll
  for (int u = 0; u < 2; ++u) {
    const int t = t0 + u, mi = t >> 2, ni = t & 3;
#pragma unroll
    for (int e = 0; e < 8; ++e) {
      const int m = mi * 16 + e + 8 * (lane >> 4);
      const int n = ni * 16 + (lane & 15);
      kvout[(size_t)bh * (DKH * DKH) + m * DKH + n] = acc[u][e];
    }
  }
  sred[tid] = ksl;
  __syncthreads();
  for (int o = 128; o > 0; o >>= 1) {
    if (tid < o) sred[tid] += sred[tid + o];
    __syncthreads();
  }
  if (tid == 0) ksum[bh] = sred[0];
}

// ---------------------------------------------------------------------------
// out[b,n,h,:] = (Q[b,n,h,:] @ kv[b,h]) / (Q.sum(-1)*ksum[b,h] + 1e-6), bf16.
// Grid: (Ntok/64, B*H); 64-token tile per block; kv resident in LDS.
// Q tile is a pure copy -> async staged; kv transpose stays direct.
// ---------------------------------------------------------------------------
__global__ __launch_bounds__(256) void linattn_out_kernel(
    const __bf16* __restrict__ Q, const float* __restrict__ kv,
    const float* __restrict__ ksum, __bf16* __restrict__ outB,
    int Ntok, int Dd) {
  __shared__ __align__(16) __bf16 Qs[64 * 64];   // [tok][d]
  __shared__ __align__(16) __bf16 kvT[64 * 64];  // [e][d]
  __shared__ float qs[64];
  const int tid = threadIdx.x, lane = tid & 31, wave = tid >> 5;
  const int bh = blockIdx.y, b = bh >> 4, h = bh & 15;
  const int nt = blockIdx.x * 64;
  const __bf16* Qb = Q + (size_t)b * Ntok * Dd + h * DKH;
  {
    const int r = tid >> 2, c = (tid & 3) * 16;
    stage16(Qs + r * 64 + c,     Qb + (size_t)(nt + r) * Dd + c);
    stage16(Qs + r * 64 + c + 8, Qb + (size_t)(nt + r) * Dd + c + 8);
  }
  {
    const float* kvb = kv + (size_t)bh * (DKH * DKH);
    const int d0 = tid >> 2, e0 = (tid & 3) * 16;
#pragma unroll
    for (int e = 0; e < 16; ++e)
      kvT[(e0 + e) * 64 + d0] = f2bf(kvb[d0 * 64 + e0 + e]);
  }
  stage_fence();
  __syncthreads();
  if (tid < 64) {
    float s = 0.0f;
    for (int i = 0; i < 64; ++i) s += bf2f(Qs[tid * 64 + i]);
    qs[tid] = s;
  }
  __syncthreads();
  const int t0 = wave * 2;
  v8f acc[2];
#pragma unroll
  for (int u = 0; u < 2; ++u)
#pragma unroll
    for (int e = 0; e < 8; ++e) acc[u][e] = 0.0f;
#pragma unroll
  for (int kk = 0; kk < 64; kk += 32) {
#pragma unroll
    for (int u = 0; u < 2; ++u) {
      const int t = t0 + u, mi = t >> 2, ni = t & 3;
      v16bf a  = load_a_frag(Qs, 64, mi * 16, kk, lane);
      v16bf bb = load_b_frag(kvT, 64, ni * 16, kk, lane);
      acc[u] = wmma_bf16(a, bb, acc[u]);
    }
  }
  const float ks = ksum[bh];
#pragma unroll
  for (int u = 0; u < 2; ++u) {
    const int t = t0 + u, mi = t >> 2, ni = t & 3;
#pragma unroll
    for (int e = 0; e < 8; ++e) {
      const int m = mi * 16 + e + 8 * (lane >> 4);
      const int n = ni * 16 + (lane & 15);
      const float sc = 1.0f / (qs[m] * ks + 1e-6f);
      outB[((size_t)b * Ntok + nt + m) * Dd + h * DKH + n] = f2bf(acc[u][e] * sc);
    }
  }
}

// ---------------------------------------------------------------------------
// Host orchestration
// ---------------------------------------------------------------------------
extern "C" void kernel_launch(void* const* d_in, const int* in_sizes, int n_in,
                              void* d_out, int out_size, void* d_ws, size_t ws_size,
                              hipStream_t stream) {
  (void)in_sizes; (void)n_in; (void)out_size; (void)ws_size;
  constexpr int D = 1024, DFF = 4096, Bv = 4, S = 4096, Lq = 512;
  constexpr int BS = Bv * S;   // 16384 tokens (self)
  constexpr int BL = Bv * Lq;  // 2048 tokens (lto)

  const float* x     = (const float*)d_in[0];
  const float* lto   = (const float*)d_in[1];
  // d_in[2], d_in[3]: masks, unused by the reference math
  const float* sa_wq = (const float*)d_in[4];
  const float* sa_wk = (const float*)d_in[5];
  const float* sa_wv = (const float*)d_in[6];
  const float* sa_wo = (const float*)d_in[7];
  const float* ca_wq = (const float*)d_in[8];
  const float* ca_wk = (const float*)d_in[9];
  const float* ca_wv = (const float*)d_in[10];
  const float* ca_wo = (const float*)d_in[11];
  const float* ff_w1 = (const float*)d_in[12];
  const float* ff_b1 = (const float*)d_in[13];
  const float* ff_w2 = (const float*)d_in[14];
  const float* ff_b2 = (const float*)d_in[15];
  const float* ln1_a = (const float*)d_in[16];
  const float* ln1_b = (const float*)d_in[17];
  const float* ln2_a = (const float*)d_in[18];
  const float* ln2_b = (const float*)d_in[19];
  const float* ln3_a = (const float*)d_in[20];
  const float* ln3_b = (const float*)d_in[21];

  char* ws = (char*)d_ws;
  size_t off = 0;
  auto take = [&](size_t bytes) -> char* {
    char* p = ws + off;
    off = (off + bytes + 255) & ~(size_t)255;
    return p;
  };
  const size_t EL = 1024ull * 1024ull;  // elems in one DxD weight
  __bf16* wbf   = (__bf16*)take(16 * EL * 2);          // all weights, bf16, TRANSPOSED [N,K]
  __bf16* xn    = (__bf16*)take((size_t)BS * D * 2);   // LN out / attn pre-wo / cross-V
  __bf16* bufA  = (__bf16*)take((size_t)BS * D * 2);   // self-Q / cross-K / FFN hidden
  __bf16* bufB  = (__bf16*)take((size_t)BS * D * 2);   // self-K / x_upd bf16
  __bf16* bufC  = (__bf16*)take((size_t)BS * D * 2);   // self-V / lto-side small bufs
  float*  xupd  = (float*)take((size_t)BS * D * 4);    // x + selfattn (f32)
  float*  kvbuf = (float*)take((size_t)Bv * HEADS * DKH * DKH * 4);
  float*  ksum  = (float*)take((size_t)Bv * HEADS * 4);
  float*  ltoupd = (float*)take((size_t)BL * D * 4);   // lto + crossattn (f32)

  __bf16* w_sq = wbf + 0 * EL;
  __bf16* w_sk = wbf + 1 * EL;
  __bf16* w_sv = wbf + 2 * EL;
  __bf16* w_so = wbf + 3 * EL;
  __bf16* w_cq = wbf + 4 * EL;
  __bf16* w_ck = wbf + 5 * EL;
  __bf16* w_cv = wbf + 6 * EL;
  __bf16* w_co = wbf + 7 * EL;
  __bf16* w_f1 = wbf + 8 * EL;    // [DFF, D]  (transposed)
  __bf16* w_f2 = wbf + 12 * EL;   // [D, DFF]  (transposed)

  __bf16* lton  = bufC;            // [BL, D]
  __bf16* Qc    = bufC + 2 * EL;   // [BL, D]
  __bf16* attnc = bufC + 4 * EL;   // [BL, D]
  __bf16* fn    = bufC + 6 * EL;   // [BL, D]

  // weight prep: W[K,N] -> WT[N,K] bf16
  auto convT = [&](const float* src, __bf16* dst, int K, int N) {
    transpose_f32_to_bf16_kernel<<<dim3(N / 32, K / 32), 256, 0, stream>>>(
        src, dst, K, N);
  };
  convT(sa_wq, w_sq, D, D); convT(sa_wk, w_sk, D, D);
  convT(sa_wv, w_sv, D, D); convT(sa_wo, w_so, D, D);
  convT(ca_wq, w_cq, D, D); convT(ca_wk, w_ck, D, D);
  convT(ca_wv, w_cv, D, D); convT(ca_wo, w_co, D, D);
  convT(ff_w1, w_f1, D, DFF); convT(ff_w2, w_f2, DFF, D);

  const dim3 g_sd(D / 128, BS / 64);   // [BS, D] GEMMs
  const dim3 g_ld(D / 128, BL / 64);   // [BL, D] GEMMs

  // ---- residual 0: self attention ----
  ln_bf16_kernel<<<BS, 256, 0, stream>>>(x, ln1_a, ln1_b, xn, D);
  gemm_bf16_kernel<<<g_sd, 256, 0, stream>>>(xn, w_sq, nullptr, nullptr, nullptr, bufA, BS, D, D, 1);
  gemm_bf16_kernel<<<g_sd, 256, 0, stream>>>(xn, w_sk, nullptr, nullptr, nullptr, bufB, BS, D, D, 1);
  gemm_bf16_kernel<<<g_sd, 256, 0, stream>>>(xn, w_sv, nullptr, nullptr, nullptr, bufC, BS, D, D, 0);
  linattn_kv_kernel<<<Bv * HEADS, 256, 0, stream>>>(bufB, bufC, kvbuf, ksum, S, D);
  linattn_out_kernel<<<dim3(S / 64, Bv * HEADS), 256, 0, stream>>>(bufA, kvbuf, ksum, xn, S, D);
  gemm_bf16_kernel<<<g_sd, 256, 0, stream>>>(xn, w_so, nullptr, x, xupd, bufB, BS, D, D, 3);

  // ---- residual 1: cross attention (Q from normed lto, K/V from updated x) ----
  ln_bf16_kernel<<<BL, 256, 0, stream>>>(lto, ln2_a, ln2_b, lton, D);
  gemm_bf16_kernel<<<g_ld, 256, 0, stream>>>(lton, w_cq, nullptr, nullptr, nullptr, Qc, BL, D, D, 1);
  gemm_bf16_kernel<<<g_sd, 256, 0, stream>>>(bufB, w_ck, nullptr, nullptr, nullptr, bufA, BS, D, D, 1);
  gemm_bf16_kernel<<<g_sd, 256, 0, stream>>>(bufB, w_cv, nullptr, nullptr, nullptr, xn, BS, D, D, 0);
  linattn_kv_kernel<<<Bv * HEADS, 256, 0, stream>>>(bufA, xn, kvbuf, ksum, S, D);
  linattn_out_kernel<<<dim3(Lq / 64, Bv * HEADS), 256, 0, stream>>>(Qc, kvbuf, ksum, attnc, Lq, D);
  gemm_bf16_kernel<<<g_ld, 256, 0, stream>>>(attnc, w_co, nullptr, lto, ltoupd, nullptr, BL, D, D, 3);

  // ---- residual 2: feed forward ----
  ln_bf16_kernel<<<BL, 256, 0, stream>>>(ltoupd, ln3_a, ln3_b, fn, D);
  gemm_bf16_kernel<<<dim3(DFF / 128, BL / 64), 256, 0, stream>>>(
      fn, w_f1, ff_b1, nullptr, nullptr, bufA, BL, DFF, D, 2);
  gemm_bf16_kernel<<<dim3(D / 128, BL / 64), 256, 0, stream>>>(
      bufA, w_f2, ff_b2, ltoupd, (float*)d_out, nullptr, BL, D, DFF, 4);
}
